// EventTransformer_80444737454145
// MI455X (gfx1250) — compile-verified, matrix operand order checked by reading, not run on hardware
//
#include <hip/hip_runtime.h>

// ---------------- problem constants ----------------
constexpr int N_DOM = 32768, D_IN = 128, D = 256, NHEAD = 4, DH = 64;
constexpr int NL = 4, DFF = 1024, BATCH = 64, SEQ = 672;
constexpr int NTOK = BATCH * SEQ;                 // 43008
constexpr int QTILES = SEQ / 16;                  // 42
constexpr int KCHUNKS = SEQ / 32;                 // 21

typedef __attribute__((ext_vector_type(16))) __bf16 v16bf;
typedef __attribute__((ext_vector_type(8)))  float  v8f;

__device__ __forceinline__ v8f wmma_bf16(v16bf a, v16bf b, v8f c) {
  return __builtin_amdgcn_wmma_f32_16x16x32_bf16(false, a, false, b, (short)0, c,
                                                 false, false);
}

// Contiguous-fragment loader for 16-bit WMMA A/B operands (ISA 7.12.2):
// per lane, elements 0..7 live at k = 8*half+0..7 and elements 8..15 at
// k = 16+8*half+0..7 -> two aligned 128-bit loads.
__device__ __forceinline__ v16bf load_frag(const __bf16* base, int half) {
  union { v16bf v; uint4 q[2]; } f;
  f.q[0] = *(const uint4*)(base + 8 * half);
  f.q[1] = *(const uint4*)(base + 16 + 8 * half);
  return f.v;
}

// ---------------- setup kernels ----------------
// transpose-convert f32 (K x N) -> bf16 (N x K)
__global__ void convT_kernel(const float* __restrict__ w, __bf16* __restrict__ wt,
                             int K, int N) {
  int i = blockIdx.x * 256 + threadIdx.x;
  if (i >= K * N) return;
  int k = i / N, n = i % N;
  wt[(size_t)n * K + k] = (__bf16)w[i];
}

__global__ void conv_kernel(const float* __restrict__ x, __bf16* __restrict__ y, int n) {
  int i = blockIdx.x * 256 + threadIdx.x;
  if (i < n) y[i] = (__bf16)x[i];
}

// deterministic per-event rank; rowmap[j] = b*SEQ + rank+1 (or -1 if overflow)
__global__ void pos_kernel(const int* __restrict__ idx, int* __restrict__ rowmap) {
  int b = threadIdx.x;
  if (b >= BATCH) return;
  int cnt = 0;
  for (int j = 0; j < N_DOM; ++j) {
    if (idx[j] == b) {
      int p = cnt + 1;
      rowmap[j] = (p < SEQ) ? (b * SEQ + p) : -1;
      ++cnt;
    }
  }
}

// h = 0 everywhere; cls token at t==0
__global__ void init_h_kernel(float* __restrict__ h, const float* __restrict__ cls) {
  size_t i = (size_t)blockIdx.x * 256 + threadIdx.x;
  if (i >= (size_t)NTOK * D) return;
  int c = (int)(i & (D - 1));
  int tok = (int)(i >> 8);
  h[i] = (tok % SEQ == 0) ? cls[c] : 0.f;
}

__global__ void init_kb_kernel(float* __restrict__ kb) {
  int i = blockIdx.x * 256 + threadIdx.x;
  if (i >= NTOK) return;
  kb[i] = (i % SEQ == 0) ? 0.f : -1e30f;
}

__global__ void unmask_kernel(float* __restrict__ kb, const int* __restrict__ rowmap) {
  int j = blockIdx.x * 256 + threadIdx.x;
  if (j >= N_DOM) return;
  int r = rowmap[j];
  if (r >= 0) kb[r] = 0.f;
}

// ---------------- generic WMMA GEMM (32x64 tile per wave) ----------------
// A: M x K bf16 row-major.  Bt: N x K bf16 (B transposed).  bias: f32[N].
// EPI 0: store bf16; 1: relu+store bf16; 2: f32 C += v; 3: f32 scatter (rowmap)
// EPI 4: qkv mode -- cols < 2D stored to C, V cols stored transposed into C2
//        as vT[b*D + (h*DH+dim)][key]  (key contiguous).
template <int EPI>
__global__ __launch_bounds__(256) void gemm_bf16_wmma(
    const __bf16* __restrict__ A, const __bf16* __restrict__ Bt,
    const float* __restrict__ bias, void* __restrict__ C, void* __restrict__ C2,
    const int* __restrict__ rowmap, int M, int N, int K) {
  int lane = threadIdx.x & 31;
  int wid = (int)((blockIdx.x * blockDim.x + threadIdx.x) >> 5);
  int ngroups = N >> 6;
  int mt = wid / ngroups, ng = wid % ngroups;
  if (mt * 32 >= M) return;
  int half = lane >> 4, mr = lane & 15;

  v8f acc[2][4];
#pragma unroll
  for (int u = 0; u < 2; ++u)
#pragma unroll
    for (int t = 0; t < 4; ++t) acc[u][t] = (v8f){0, 0, 0, 0, 0, 0, 0, 0};

  const __bf16* arow0 = A + (size_t)(mt * 32 + mr) * K;
  const __bf16* arow1 = arow0 + (size_t)16 * K;
  const __bf16* brow0 = Bt + (size_t)(ng * 64 + mr) * K;
  for (int kk = 0; kk < K; kk += 32) {
    v16bf a0 = load_frag(arow0 + kk, half);
    v16bf a1 = load_frag(arow1 + kk, half);
#pragma unroll
    for (int t = 0; t < 4; ++t) {
      v16bf b = load_frag(brow0 + (size_t)t * 16 * K + kk, half);
      acc[0][t] = wmma_bf16(a0, b, acc[0][t]);
      acc[1][t] = wmma_bf16(a1, b, acc[1][t]);
    }
  }

#pragma unroll
  for (int u = 0; u < 2; ++u) {
#pragma unroll
    for (int t = 0; t < 4; ++t) {
#pragma unroll
      for (int r = 0; r < 8; ++r) {
        int row = mt * 32 + u * 16 + r + 8 * half;
        int col = ng * 64 + t * 16 + mr;
        float v = acc[u][t][r] + bias[col];
        if (EPI == 0) {
          ((__bf16*)C)[(size_t)row * N + col] = (__bf16)v;
        } else if (EPI == 1) {
          ((__bf16*)C)[(size_t)row * N + col] = (__bf16)fmaxf(v, 0.f);
        } else if (EPI == 2) {
          ((float*)C)[(size_t)row * N + col] += v;
        } else if (EPI == 3) {
          int dr = rowmap[row];
          if (dr >= 0) ((float*)C)[(size_t)dr * N + col] = v;
        } else {  // EPI == 4
          if (col < 2 * D) {
            ((__bf16*)C)[(size_t)row * N + col] = (__bf16)v;
          } else {
            int hd = col - 2 * D;            // h*DH + dim, 0..255
            int bb = row / SEQ, key = row % SEQ;
            ((__bf16*)C2)[((size_t)bb * D + hd) * SEQ + key] = (__bf16)v;
          }
        }
      }
    }
  }
}

// ---------------- layernorm (one wave per token, D=256) ----------------
__global__ __launch_bounds__(256) void ln_kernel(const float* __restrict__ h,
                                                 const float* __restrict__ g,
                                                 const float* __restrict__ bb,
                                                 __bf16* __restrict__ y) {
  int lane = threadIdx.x & 31;
  int tok = (int)((blockIdx.x * blockDim.x + threadIdx.x) >> 5);
  if (tok >= NTOK) return;
  const float* x = h + (size_t)tok * D;
  float v[8], s = 0.f, s2 = 0.f;
#pragma unroll
  for (int i = 0; i < 8; ++i) {
    float t = x[lane * 8 + i];
    v[i] = t; s += t; s2 += t * t;
  }
#pragma unroll
  for (int off = 16; off > 0; off >>= 1) {
    s += __shfl_xor(s, off, 32);
    s2 += __shfl_xor(s2, off, 32);
  }
  float mean = s * (1.f / D);
  float var = s2 * (1.f / D) - mean * mean;
  float rstd = rsqrtf(var + 1e-5f);
  __bf16* yo = y + (size_t)tok * D;
#pragma unroll
  for (int i = 0; i < 8; ++i) {
    int c = lane * 8 + i;
    yo[c] = (__bf16)((v[i] - mean) * rstd * g[c] + bb[c]);
  }
}

// ---------------- flash attention: one wave per (b, head, 16-q tile) ----------------
// qkv: (NTOK, 768) bf16 (only q,k columns used).  vT: (BATCH*D, SEQ) bf16.
__global__ __launch_bounds__(256) void attn_kernel(const __bf16* __restrict__ qkv,
                                                   const __bf16* __restrict__ vT,
                                                   const float* __restrict__ kbias,
                                                   __bf16* __restrict__ aout) {
  __shared__ __align__(16) float S[8][16][32];
  __shared__ __align__(16) __bf16 Pl[8][16][32];
  __shared__ float stM[8][16], stL[8][16], stC[8][16];
  int lane = threadIdx.x & 31, wv = threadIdx.x >> 5;
  int wid = blockIdx.x * 8 + wv;
  if (wid >= BATCH * NHEAD * QTILES) return;
  int qt = wid % QTILES;
  int t = wid / QTILES;
  int hh = t & 3, b = t >> 2;
  int half = lane >> 4, mr = lane & 15;

  const __bf16* qb = qkv + (size_t)(b * SEQ + qt * 16 + mr) * (3 * D) + hh * DH;
  v16bf qa0 = load_frag(qb, half);
  v16bf qa1 = load_frag(qb + 32, half);
  if (lane < 16) { stM[wv][lane] = -1e30f; stL[wv][lane] = 0.f; }

  v8f o[4];
#pragma unroll
  for (int nt = 0; nt < 4; ++nt) o[nt] = (v8f){0, 0, 0, 0, 0, 0, 0, 0};

  for (int kc = 0; kc < KCHUNKS; ++kc) {
    int key0 = kc * 32;
    const __bf16* kb = qkv + (size_t)(b * SEQ + key0 + mr) * (3 * D) + D + hh * DH;
    v16bf kf00 = load_frag(kb, half);
    v16bf kf01 = load_frag(kb + 32, half);
    v16bf kf10 = load_frag(kb + (size_t)16 * (3 * D), half);
    v16bf kf11 = load_frag(kb + (size_t)16 * (3 * D) + 32, half);

    v8f zc = {0, 0, 0, 0, 0, 0, 0, 0};
    v8f s0 = wmma_bf16(qa0, kf00, zc);
    s0 = wmma_bf16(qa1, kf01, s0);
    v8f s1 = wmma_bf16(qa0, kf10, zc);
    s1 = wmma_bf16(qa1, kf11, s1);

    float b0 = kbias[b * SEQ + key0 + mr];
    float b1 = kbias[b * SEQ + key0 + 16 + mr];
#pragma unroll
    for (int r = 0; r < 8; ++r) {
      int row = r + 8 * half;
      S[wv][row][mr] = s0[r] * 0.125f + b0;
      S[wv][row][16 + mr] = s1[r] * 0.125f + b1;
    }
    __builtin_amdgcn_wave_barrier();
    if (lane < 16) {
      float mo = stM[wv][lane], mx = mo;
#pragma unroll
      for (int c = 0; c < 32; ++c) mx = fmaxf(mx, S[wv][lane][c]);
      float corr = __expf(mo - mx);
      float sum = 0.f;
#pragma unroll
      for (int c = 0; c < 32; ++c) {
        float p = __expf(S[wv][lane][c] - mx);
        Pl[wv][lane][c] = (__bf16)p;
        sum += p;
      }
      stM[wv][lane] = mx;
      stL[wv][lane] = stL[wv][lane] * corr + sum;
      stC[wv][lane] = corr;
    }
    __builtin_amdgcn_wave_barrier();
#pragma unroll
    for (int r = 0; r < 8; ++r) {
      float c = stC[wv][r + 8 * half];
      o[0][r] *= c; o[1][r] *= c; o[2][r] *= c; o[3][r] *= c;
    }
    v16bf pf = load_frag(&Pl[wv][mr][0], half);
    const __bf16* vb = vT + ((size_t)b * D + hh * DH + mr) * SEQ + key0;
#pragma unroll
    for (int nt = 0; nt < 4; ++nt) {
      v16bf vf = load_frag(vb + (size_t)nt * 16 * SEQ, half);
      o[nt] = wmma_bf16(pf, vf, o[nt]);
    }
  }

#pragma unroll
  for (int nt = 0; nt < 4; ++nt) {
#pragma unroll
    for (int r = 0; r < 8; ++r) {
      int row = r + 8 * half;
      float inv = 1.f / stL[wv][row];
      aout[(size_t)(b * SEQ + qt * 16 + row) * D + hh * DH + nt * 16 + mr] =
          (__bf16)(o[nt][r] * inv);
    }
  }
}

// ---------------- head ----------------
__global__ void gather_cls_kernel(const float* __restrict__ h, __bf16* __restrict__ cls) {
  int i = blockIdx.x * 256 + threadIdx.x;
  if (i >= BATCH * D) return;
  int b = i >> 8, c = i & 255;
  cls[i] = (__bf16)h[(size_t)(b * SEQ) * D + c];
}

__global__ void head_final_kernel(const __bf16* __restrict__ hm,
                                  const float* __restrict__ w2,
                                  const float* __restrict__ b2,
                                  float* __restrict__ out) {
  int b = threadIdx.x;
  if (b >= BATCH) return;
  float p0 = b2[0], p1 = b2[1], p2 = b2[2];
  for (int k = 0; k < D; ++k) {
    float v = (float)hm[b * D + k];
    p0 += v * w2[k * 3 + 0];
    p1 += v * w2[k * 3 + 1];
    p2 += v * w2[k * 3 + 2];
  }
  float inv = rsqrtf(p0 * p0 + p1 * p1 + p2 * p2);
  out[b * 3 + 0] = p0 * inv;
  out[b * 3 + 1] = p1 * inv;
  out[b * 3 + 2] = p2 * inv;
}

// ---------------- host ----------------
static inline size_t alignup(size_t x) { return (x + 255) & ~(size_t)255; }

extern "C" void kernel_launch(void* const* d_in, const int* in_sizes, int n_in,
                              void* d_out, int out_size, void* d_ws, size_t ws_size,
                              hipStream_t stream) {
  const float* dom    = (const float*)d_in[0];
  const int*   idx    = (const int*)d_in[1];
  const float* proj_w = (const float*)d_in[4];
  const float* proj_b = (const float*)d_in[5];
  const float* cls_tk = (const float*)d_in[6];
  const float* ln1_g  = (const float*)d_in[7];
  const float* ln1_b  = (const float*)d_in[8];
  const float* qkv_w  = (const float*)d_in[9];
  const float* qkv_b  = (const float*)d_in[10];
  const float* out_w  = (const float*)d_in[11];
  const float* out_b  = (const float*)d_in[12];
  const float* ln2_g  = (const float*)d_in[13];
  const float* ln2_b  = (const float*)d_in[14];
  const float* ff1_w  = (const float*)d_in[15];
  const float* ff1_b  = (const float*)d_in[16];
  const float* ff2_w  = (const float*)d_in[17];
  const float* ff2_b  = (const float*)d_in[18];
  const float* h1_w   = (const float*)d_in[19];
  const float* h1_b   = (const float*)d_in[20];
  const float* h2_w   = (const float*)d_in[21];
  const float* h2_b   = (const float*)d_in[22];

  // workspace layout
  char* ws = (char*)d_ws;
  size_t off = 0;
  float*  h       = (float*)(ws + off);  off = alignup(off + (size_t)NTOK * D * 4);
  __bf16* y       = (__bf16*)(ws + off); off = alignup(off + (size_t)NTOK * D * 2);
  __bf16* big     = (__bf16*)(ws + off); off = alignup(off + (size_t)NTOK * DFF * 2); // qkv / ff_mid
  __bf16* vT      = (__bf16*)(ws + off); off = alignup(off + (size_t)NTOK * D * 2);
  __bf16* aout    = (__bf16*)(ws + off); off = alignup(off + (size_t)NTOK * D * 2);
  float*  kbf     = (float*)(ws + off);  off = alignup(off + (size_t)NTOK * 4);
  int*    rowmap  = (int*)(ws + off);    off = alignup(off + (size_t)N_DOM * 4);
  __bf16* domA    = (__bf16*)(ws + off); off = alignup(off + (size_t)N_DOM * D_IN * 2);
  __bf16* proj_wT = (__bf16*)(ws + off); off = alignup(off + (size_t)D_IN * D * 2);
  __bf16* qkv_wT  = (__bf16*)(ws + off); off = alignup(off + (size_t)NL * D * 3 * D * 2);
  __bf16* out_wT  = (__bf16*)(ws + off); off = alignup(off + (size_t)NL * D * D * 2);
  __bf16* ff1_wT  = (__bf16*)(ws + off); off = alignup(off + (size_t)NL * D * DFF * 2);
  __bf16* ff2_wT  = (__bf16*)(ws + off); off = alignup(off + (size_t)NL * DFF * D * 2);
  __bf16* h1_wT   = (__bf16*)(ws + off); off = alignup(off + (size_t)D * D * 2);
  __bf16* clsb    = (__bf16*)(ws + off); off = alignup(off + (size_t)BATCH * D * 2);
  __bf16* hm      = (__bf16*)(ws + off); off = alignup(off + (size_t)BATCH * D * 2);
  if (off > ws_size) return;  // insufficient scratch

  auto cdiv = [](size_t a, size_t b) { return (int)((a + b - 1) / b); };
  auto gemm_blocks = [&](int M, int N) { return cdiv((size_t)(M / 32) * (N / 64) * 32, 256); };

  // --- weight conversion (transposed bf16) ---
  convT_kernel<<<cdiv(D_IN * D, 256), 256, 0, stream>>>(proj_w, proj_wT, D_IN, D);
  for (int i = 0; i < NL; ++i) {
    convT_kernel<<<cdiv(D * 3 * D, 256), 256, 0, stream>>>(
        qkv_w + (size_t)i * D * 3 * D, qkv_wT + (size_t)i * 3 * D * D, D, 3 * D);
    convT_kernel<<<cdiv(D * D, 256), 256, 0, stream>>>(
        out_w + (size_t)i * D * D, out_wT + (size_t)i * D * D, D, D);
    convT_kernel<<<cdiv(D * DFF, 256), 256, 0, stream>>>(
        ff1_w + (size_t)i * D * DFF, ff1_wT + (size_t)i * D * DFF, D, DFF);
    convT_kernel<<<cdiv(DFF * D, 256), 256, 0, stream>>>(
        ff2_w + (size_t)i * DFF * D, ff2_wT + (size_t)i * DFF * D, DFF, D);
  }
  convT_kernel<<<cdiv(D * D, 256), 256, 0, stream>>>(h1_w, h1_wT, D, D);
  conv_kernel<<<cdiv((size_t)N_DOM * D_IN, 256), 256, 0, stream>>>(dom, domA, N_DOM * D_IN);

  // --- scatter prep ---
  pos_kernel<<<1, 64, 0, stream>>>(idx, rowmap);
  init_h_kernel<<<cdiv((size_t)NTOK * D, 256), 256, 0, stream>>>(h, cls_tk);
  init_kb_kernel<<<cdiv(NTOK, 256), 256, 0, stream>>>(kbf);
  unmask_kernel<<<cdiv(N_DOM, 256), 256, 0, stream>>>(kbf, rowmap);

  // --- projection GEMM fused with scatter into h ---
  gemm_bf16_wmma<3><<<gemm_blocks(N_DOM, D), 256, 0, stream>>>(
      domA, proj_wT, proj_b, (void*)h, nullptr, rowmap, N_DOM, D, D_IN);

  // --- transformer layers ---
  for (int i = 0; i < NL; ++i) {
    ln_kernel<<<cdiv((size_t)NTOK * 32, 256), 256, 0, stream>>>(
        h, ln1_g + i * D, ln1_b + i * D, y);
    gemm_bf16_wmma<4><<<gemm_blocks(NTOK, 3 * D), 256, 0, stream>>>(
        y, qkv_wT + (size_t)i * 3 * D * D, qkv_b + i * 3 * D, (void*)big, (void*)vT,
        nullptr, NTOK, 3 * D, D);
    attn_kernel<<<cdiv((size_t)BATCH * NHEAD * QTILES * 32, 256), 256, 0, stream>>>(
        big, vT, kbf, aout);
    gemm_bf16_wmma<2><<<gemm_blocks(NTOK, D), 256, 0, stream>>>(
        aout, out_wT + (size_t)i * D * D, out_b + i * D, (void*)h, nullptr, nullptr,
        NTOK, D, D);
    ln_kernel<<<cdiv((size_t)NTOK * 32, 256), 256, 0, stream>>>(
        h, ln2_g + i * D, ln2_b + i * D, y);
    gemm_bf16_wmma<1><<<gemm_blocks(NTOK, DFF), 256, 0, stream>>>(
        y, ff1_wT + (size_t)i * D * DFF, ff1_b + i * DFF, (void*)big, nullptr, nullptr,
        NTOK, DFF, D);
    gemm_bf16_wmma<2><<<gemm_blocks(NTOK, D), 256, 0, stream>>>(
        big, ff2_wT + (size_t)i * DFF * D, ff2_b + i * D, (void*)h, nullptr, nullptr,
        NTOK, D, DFF);
  }

  // --- head ---
  gather_cls_kernel<<<cdiv(BATCH * D, 256), 256, 0, stream>>>(h, clsb);
  gemm_bf16_wmma<1><<<gemm_blocks(BATCH, D), 256, 0, stream>>>(
      clsb, h1_wT, h1_b, (void*)hm, nullptr, nullptr, BATCH, D, D);
  head_final_kernel<<<1, 64, 0, stream>>>(hm, h2_w, h2_b, (float*)d_out);
}
